// attn_head_87531433492692
// MI455X (gfx1250) — matrix operand order repaired and checked by reading, not compiled, not run on hardware
//
#include <hip/hip_runtime.h>
#include <hip/hip_bf16.h>

// GAT attention head, fused flash-style for gfx1250 (MI455X).
// N=8192, F_in=256, F_out=64.
//
// Kernel 1 (proj): fts = x@W (f32), store ftsT as f16 [64][8192]; f1=fts@a1+b1, f2=fts@a2+b2.
// Kernel 2 (gat):  per 16-row tile, stream 8192 cols in K=32 chunks (256MB adj read once,
//                  non-temporal); online softmax with branch-skipped rescale; accumulate
//                  P(16x32,f16) x ftsT(32x16,f16) via v_wmma_f32_16x16x32_f16 into 4 N-tiles.

typedef __attribute__((ext_vector_type(16))) _Float16 v16h;
typedef __attribute__((ext_vector_type(8)))  float    v8f;
typedef __attribute__((ext_vector_type(4)))  float    v4f;

#define GAT_N    8192
#define GAT_FIN  256
#define GAT_FOUT 64
#define LRELU_A  0.2f

__device__ __forceinline__ float lane_bcast(float v, int srcLane) {
    return __int_as_float(__builtin_amdgcn_ds_bpermute(srcLane << 2, __float_as_int(v)));
}

// ---------------- Kernel 1: projection + attention logit pieces ----------------
__global__ __launch_bounds__(64) void gat_proj_kernel(
    const float* __restrict__ x, const float* __restrict__ W,
    const float* __restrict__ a1, const float* __restrict__ b1,
    const float* __restrict__ a2, const float* __restrict__ b2,
    _Float16* __restrict__ ftsT, float* __restrict__ f1, float* __restrict__ f2)
{
    const int n   = blockIdx.x;   // node
    const int tid = threadIdx.x;  // output feature 0..63
    __shared__ float xs[GAT_FIN];
    __shared__ float red1[GAT_FOUT];
    __shared__ float red2[GAT_FOUT];

    for (int k = tid; k < GAT_FIN; k += GAT_FOUT)
        xs[k] = x[(size_t)n * GAT_FIN + k];
    __syncthreads();

    float acc = 0.f;
    #pragma unroll 8
    for (int k = 0; k < GAT_FIN; ++k)
        acc = fmaf(xs[k], W[k * GAT_FOUT + tid], acc);

    // transposed f16 copy of fts for the WMMA B-matrix in kernel 2
    ftsT[(size_t)tid * GAT_N + n] = (_Float16)acc;

    red1[tid] = acc * a1[tid];
    red2[tid] = acc * a2[tid];
    __syncthreads();
    for (int s = GAT_FOUT / 2; s > 0; s >>= 1) {
        if (tid < s) { red1[tid] += red1[tid + s]; red2[tid] += red2[tid + s]; }
        __syncthreads();
    }
    if (tid == 0) {
        f1[n] = red1[0] + b1[0];
        f2[n] = red2[0] + b2[0];
    }
}

// ---------------- Kernel 2: fused online-softmax attention ----------------
__global__ __launch_bounds__(32) void gat_attn_kernel(
    const float* __restrict__ adj, const float* __restrict__ f1v,
    const float* __restrict__ f2v, const _Float16* __restrict__ ftsT,
    const float* __restrict__ bias, float* __restrict__ out)
{
    const int lane = threadIdx.x;      // 0..31 (wave32)
    const int half = lane >> 4;        // which K-half this lane owns
    const int lrow = lane & 15;        // tile row this lane produces A/stats for
    const int r0   = blockIdx.x * 16;
    const int myrow = r0 + lrow;

    const float f1r = f1v[myrow];
    const float* __restrict__ adjrow = adj + (size_t)myrow * GAT_N;

    float M = -1e30f;   // running row max (identical sequence in lanes l and l+16)
    float S = 0.f;      // PARTIAL (per K-half) running sum of exp; combined at the end
    v8f c0 = {}, c1 = {}, c2 = {}, c3 = {};

    float bia[4];
    #pragma unroll
    for (int t = 0; t < 4; ++t) bia[t] = bias[t * 16 + lrow];

    for (int cb = 0; cb < GAT_N; cb += 32) {
        // A-matrix K layout (16-bit, 16x32): lanes 0-15 own K {0..15},
        // lanes 16-31 own K {16..31} of this chunk.
        const int klo = cb + 16 * half;
        const int khi = cb + 16 * half + 8;

        // Streaming prefetch ~2KB ahead on this lane's adj row.
        if (cb + 544 <= GAT_N)
            __builtin_prefetch(adjrow + klo + 512, 0, 1);

        float aa[16], gg[16];
        ((v4f*)aa)[0] = __builtin_nontemporal_load((const v4f*)(adjrow + klo));
        ((v4f*)aa)[1] = __builtin_nontemporal_load((const v4f*)(adjrow + klo) + 1);
        ((v4f*)aa)[2] = __builtin_nontemporal_load((const v4f*)(adjrow + khi));
        ((v4f*)aa)[3] = __builtin_nontemporal_load((const v4f*)(adjrow + khi) + 1);
        ((v4f*)gg)[0] = *((const v4f*)(f2v + klo));
        ((v4f*)gg)[1] = *((const v4f*)(f2v + klo) + 1);
        ((v4f*)gg)[2] = *((const v4f*)(f2v + khi));
        ((v4f*)gg)[3] = *((const v4f*)(f2v + khi) + 1);

        float s[16];
        float tmax = -1e30f;
        #pragma unroll
        for (int i = 0; i < 16; ++i) {
            float t = f1r + gg[i];
            s[i] = fmaxf(t, LRELU_A * t) + aa[i];   // leaky_relu + adj bias
            tmax = fmaxf(tmax, s[i]);
        }

        // Rare path: some row's max moved -> combine halves, rescale C and S.
        if (__builtin_amdgcn_ballot_w32(tmax > M)) {
            float tfull = fmaxf(tmax, lane_bcast(tmax, lane ^ 16));
            float Mnew  = fmaxf(M, tfull);
            float scale = __expf(M - Mnew);
            S *= scale;                 // per-half partial, own-row scale
            #pragma unroll
            for (int j = 0; j < 8; ++j) {
                // C VGPR j holds row j + 8*half; row r's scale lives in lane r.
                float sj = lane_bcast(scale, j + 8 * half);
                c0[j] *= sj; c1[j] *= sj; c2[j] *= sj; c3[j] *= sj;
            }
            M = Mnew;
        }

        // Steady state: all s <= M, so exp(s-M) <= 1. Keep per-half partial sums.
        float tsum = 0.f;
        v16h pa;
        #pragma unroll
        for (int i = 0; i < 16; ++i) {
            float p = __expf(s[i] - M);
            tsum += p;
            pa[i] = (_Float16)p;
        }
        S += tsum;

        // B-matrix (32x16 f16): lane owns feature 16t+lrow, 16 consecutive K (by half).
        const _Float16* bb = ftsT + (size_t)lrow * GAT_N + cb + 16 * half;
        v16h b0 = *(const v16h*)(bb);
        v16h b1 = *(const v16h*)(bb + (size_t)16 * GAT_N);
        v16h b2 = *(const v16h*)(bb + (size_t)32 * GAT_N);
        v16h b3 = *(const v16h*)(bb + (size_t)48 * GAT_N);

        c0 = __builtin_amdgcn_wmma_f32_16x16x32_f16(false, pa, false, b0, (short)0, c0, false, false);
        c1 = __builtin_amdgcn_wmma_f32_16x16x32_f16(false, pa, false, b1, (short)0, c1, false, false);
        c2 = __builtin_amdgcn_wmma_f32_16x16x32_f16(false, pa, false, b2, (short)0, c2, false, false);
        c3 = __builtin_amdgcn_wmma_f32_16x16x32_f16(false, pa, false, b3, (short)0, c3, false, false);
    }

    // Combine the two K-half partial sums -> full softmax denominator per row.
    const float Stot = S + lane_bcast(S, lane ^ 16);
    const float Sinv = __builtin_amdgcn_rcpf(Stot);

    // Normalize, bias, ELU, store. C VGPR j, lane l: row r0+j+8*half, feature 16t+lrow.
    #pragma unroll
    for (int j = 0; j < 8; ++j) {
        float sj = lane_bcast(Sinv, j + 8 * half);
        float* __restrict__ orow = out + (size_t)(r0 + j + 8 * half) * GAT_FOUT;
        float v;
        v = c0[j] * sj + bia[0]; __builtin_nontemporal_store(v > 0.f ? v : __expf(v) - 1.f, orow +  0 + lrow);
        v = c1[j] * sj + bia[1]; __builtin_nontemporal_store(v > 0.f ? v : __expf(v) - 1.f, orow + 16 + lrow);
        v = c2[j] * sj + bia[2]; __builtin_nontemporal_store(v > 0.f ? v : __expf(v) - 1.f, orow + 32 + lrow);
        v = c3[j] * sj + bia[3]; __builtin_nontemporal_store(v > 0.f ? v : __expf(v) - 1.f, orow + 48 + lrow);
    }
}

extern "C" void kernel_launch(void* const* d_in, const int* in_sizes, int n_in,
                              void* d_out, int out_size, void* d_ws, size_t ws_size,
                              hipStream_t stream) {
    const float* x    = (const float*)d_in[0];
    const float* adj  = (const float*)d_in[1];
    const float* W    = (const float*)d_in[2];
    const float* a1   = (const float*)d_in[3];
    const float* b1   = (const float*)d_in[4];
    const float* a2   = (const float*)d_in[5];
    const float* b2   = (const float*)d_in[6];
    const float* bias = (const float*)d_in[7];
    float* out = (float*)d_out;

    // workspace: ftsT f16 [64][8192] (1 MB), f1[8192], f2[8192] (f32)
    _Float16* ftsT = (_Float16*)d_ws;
    float* f1 = (float*)((char*)d_ws + (size_t)GAT_FOUT * GAT_N * sizeof(_Float16));
    float* f2 = f1 + GAT_N;

    gat_proj_kernel<<<GAT_N, GAT_FOUT, 0, stream>>>(x, W, a1, b1, a2, b2, ftsT, f1, f2);
    gat_attn_kernel<<<GAT_N / 16, 32, 0, stream>>>(adj, f1, f2, ftsT, bias, out);
}
